// T5Attention_61074434949293
// MI455X (gfx1250) — matrix operand order repaired and checked by reading, not compiled, not run on hardware
//
#include <hip/hip_runtime.h>

// ---------------------------------------------------------------------------
// T5 attention (no 1/sqrt(d) scaling), B=2, T=2048, D=1024, H=16, Dh=64.
// bf16 WMMA (v_wmma_f32_16x16x32_bf16) everywhere, f32 accumulate,
// flash-style online softmax so the 512MB score tensor is never materialized.
// ---------------------------------------------------------------------------

#define D_MODEL 1024
#define N_HEADS 16
#define HEAD_DIM 64
#define B_SZ 2
#define T_SEQ 2048

typedef __attribute__((ext_vector_type(8)))  __bf16 v8bf;
typedef __attribute__((ext_vector_type(16))) __bf16 v16bf;
typedef __attribute__((ext_vector_type(8)))  float  v8f;

static __device__ __forceinline__ unsigned short f2bu(float f) {
    unsigned int u = __builtin_bit_cast(unsigned int, f);
    u += 0x7FFFu + ((u >> 16) & 1u);   // round-to-nearest-even
    return (unsigned short)(u >> 16);
}

// Load a 16-element bf16 fragment as two 16B segments at +0 and +off2.
static __device__ __forceinline__ v16bf load16(const __bf16* p, int off2) {
    v8bf lo = *(const v8bf*)p;
    v8bf hi = *(const v8bf*)(p + off2);
    v16bf r;
#pragma unroll
    for (int i = 0; i < 8; ++i) { r[i] = lo[i]; r[i + 8] = hi[i]; }
    return r;
}

static __device__ __forceinline__ v8f wmma_bf16(v16bf a, v16bf b, v8f c) {
    return __builtin_amdgcn_wmma_f32_16x16x32_bf16(
        /*neg_a=*/false, a, /*neg_b=*/false, b,
        /*c_mod=*/(short)0, c, /*reuse_a=*/false, /*reuse_b=*/false);
}

// ---------------------------------------------------------------------------
// f32 -> bf16 conversion
// ---------------------------------------------------------------------------
__global__ void cvt_bf16(const float* __restrict__ in,
                         unsigned short* __restrict__ out, int n) {
    int i = blockIdx.x * blockDim.x + threadIdx.x;
    if (i < n) out[i] = f2bu(in[i]);
}

// ---------------------------------------------------------------------------
// Fused QKV projection: Y = X @ W^T.  One wave per 16x16 tile, K=1024.
// blockIdx.x: o-tile (64), blockIdx.y: t-tile (256), blockIdx.z: q/k/v.
// V is written transposed [b,h,d,t] so attention PV B-frags are contiguous.
// ---------------------------------------------------------------------------
__global__ __launch_bounds__(32) void qkv_gemm(
    const unsigned short* __restrict__ xb,
    const unsigned short* __restrict__ wq,
    const unsigned short* __restrict__ wk,
    const unsigned short* __restrict__ wv,
    unsigned short* __restrict__ qo,
    unsigned short* __restrict__ ko,
    unsigned short* __restrict__ vto) {
    const int lane = threadIdx.x & 31;
    const int ln = lane & 15, half = lane >> 4;
    const int ob = blockIdx.x * 16;
    const int tb = blockIdx.y * 16;
    const int which = blockIdx.z;
    const unsigned short* w = (which == 0) ? wq : (which == 1) ? wk : wv;

    const __bf16* xrow = (const __bf16*)xb + (size_t)(tb + ln) * D_MODEL;
    const __bf16* wrow = (const __bf16*)w  + (size_t)(ob + ln) * D_MODEL;

    v8f acc = {};
#pragma unroll 4
    for (int kc = 0; kc < D_MODEL / 32; ++kc) {
        __builtin_prefetch(xrow + kc * 32 + 64, 0, 3);
        v16bf a = load16(xrow + kc * 32 + half * 16, 8);   // A 16x32
        v16bf b = load16(wrow + kc * 32 + half * 16, 8);   // B 32x16 (W^T)
        acc = wmma_bf16(a, b, acc);
    }

    if (which == 2) {
        const int o = ob + ln;
        const int h = o >> 6, d = o & 63;
#pragma unroll
        for (int r = 0; r < 8; ++r) {
            int row = tb + half * 8 + r;             // global b*T + t
            int bi = row >> 11, t = row & (T_SEQ - 1);
            vto[(size_t)((bi * N_HEADS + h) * HEAD_DIM + d) * T_SEQ + t] =
                f2bu(acc[r]);
        }
    } else {
        unsigned short* dst = (which == 0) ? qo : ko;
#pragma unroll
        for (int r = 0; r < 8; ++r) {
            int row = tb + half * 8 + r;
            dst[(size_t)row * D_MODEL + ob + ln] = f2bu(acc[r]);
        }
    }
}

// ---------------------------------------------------------------------------
// Flash attention: one wave per (q-tile of 16, head, batch).
// S = Q@K^T + bias, online softmax over 32-key blocks, O += P@V.
// ---------------------------------------------------------------------------
__global__ __launch_bounds__(32) void attn_flash(
    const unsigned short* __restrict__ qb,
    const unsigned short* __restrict__ kb,
    const unsigned short* __restrict__ vtb,
    const float* __restrict__ bias,
    unsigned short* __restrict__ ctx) {
    __shared__ unsigned short Plds[16 * 32];

    const int lane = threadIdx.x & 31;
    const int ln = lane & 15, half = lane >> 4;
    const int qt = blockIdx.x, h = blockIdx.y, b = blockIdx.z;
    const int qbse = qt * 16;

    // Q A-fragments (M=16, K=64 split into two 32-chunks)
    const __bf16* qrow = (const __bf16*)qb +
        (size_t)(b * T_SEQ + qbse + ln) * D_MODEL + h * HEAD_DIM;
    v16bf qa0 = load16(qrow + half * 16, 8);
    v16bf qa1 = load16(qrow + 32 + half * 16, 8);

    v8f o0 = {}, o1 = {}, o2 = {}, o3 = {};
    float m[8], l[8];
#pragma unroll
    for (int r = 0; r < 8; ++r) { m[r] = -1e30f; l[r] = 0.0f; }

    const __bf16* kbase = (const __bf16*)kb +
        (size_t)(b * T_SEQ) * D_MODEL + h * HEAD_DIM;
    const __bf16* vbase = (const __bf16*)vtb +
        (size_t)((b * N_HEADS + h) * HEAD_DIM) * T_SEQ;
    const float* bbase = bias +
        ((size_t)h * T_SEQ + qbse + half * 8) * T_SEQ;

    for (int kk = 0; kk < T_SEQ; kk += 32) {
        // ----- S = Q @ K^T for two 16-key tiles ------------------------------
        const __bf16* krow0 = kbase + (size_t)(kk + ln) * D_MODEL;
        const __bf16* krow1 = krow0 + (size_t)16 * D_MODEL;
        v8f s0 = {}, s1 = {};
        {
            v16bf b00 = load16(krow0 + half * 16, 8);
            v16bf b01 = load16(krow0 + 32 + half * 16, 8);
            s0 = wmma_bf16(qa0, b00, s0);
            s0 = wmma_bf16(qa1, b01, s0);
            v16bf b10 = load16(krow1 + half * 16, 8);
            v16bf b11 = load16(krow1 + 32 + half * 16, 8);
            s1 = wmma_bf16(qa0, b10, s1);
            s1 = wmma_bf16(qa1, b11, s1);
        }

        // ----- bias add + online softmax ------------------------------------
        const float* bp = bbase + kk + ln;
#pragma unroll
        for (int r = 0; r < 8; ++r) {
            float v0 = s0[r] + bp[(size_t)r * T_SEQ];
            float v1 = s1[r] + bp[(size_t)r * T_SEQ + 16];
            float rm = fmaxf(v0, v1);
#pragma unroll
            for (int off = 8; off > 0; off >>= 1)
                rm = fmaxf(rm, __shfl_xor(rm, off, 16));
            float mn = fmaxf(m[r], rm);
            float alpha = __expf(m[r] - mn);
            m[r] = mn;
            float p0 = __expf(v0 - mn);
            float p1 = __expf(v1 - mn);
            float rs = p0 + p1;
#pragma unroll
            for (int off = 8; off > 0; off >>= 1)
                rs += __shfl_xor(rs, off, 16);
            l[r] = l[r] * alpha + rs;
            o0[r] *= alpha; o1[r] *= alpha; o2[r] *= alpha; o3[r] *= alpha;
            // stage P (C-layout -> LDS, row-major 16x32 bf16)
            Plds[(half * 8 + r) * 32 + ln]      = f2bu(p0);
            Plds[(half * 8 + r) * 32 + 16 + ln] = f2bu(p1);
        }
        __syncthreads();
        // P as A-fragment (M=16, K=32 keys)
        v16bf pa = load16((const __bf16*)&Plds[ln * 32 + half * 16], 8);
        __syncthreads();

        // ----- O += P @ V (V transposed: contiguous key runs) ---------------
        {
            const __bf16* vr = vbase + (size_t)ln * T_SEQ + kk + half * 16;
            o0 = wmma_bf16(pa, load16(vr, 8), o0);
            vr += (size_t)16 * T_SEQ;
            o1 = wmma_bf16(pa, load16(vr, 8), o1);
            vr += (size_t)16 * T_SEQ;
            o2 = wmma_bf16(pa, load16(vr, 8), o2);
            vr += (size_t)16 * T_SEQ;
            o3 = wmma_bf16(pa, load16(vr, 8), o3);
        }
    }

    // ----- normalize and emit ctx[b,t,h*64+d] as bf16 ------------------------
#pragma unroll
    for (int r = 0; r < 8; ++r) {
        float inv = 1.0f / l[r];
        size_t base = (size_t)(b * T_SEQ + qbse + half * 8 + r) * D_MODEL +
                      h * HEAD_DIM + ln;
        ctx[base]      = f2bu(o0[r] * inv);
        ctx[base + 16] = f2bu(o1[r] * inv);
        ctx[base + 32] = f2bu(o2[r] * inv);
        ctx[base + 48] = f2bu(o3[r] * inv);
    }
}

// ---------------------------------------------------------------------------
// Output projection: out = ctx @ Wo^T, f32 output.
// ---------------------------------------------------------------------------
__global__ __launch_bounds__(32) void out_gemm(
    const unsigned short* __restrict__ ctx,
    const unsigned short* __restrict__ wo,
    float* __restrict__ out) {
    const int lane = threadIdx.x & 31;
    const int ln = lane & 15, half = lane >> 4;
    const int ob = blockIdx.x * 16;
    const int tb = blockIdx.y * 16;

    const __bf16* xrow = (const __bf16*)ctx + (size_t)(tb + ln) * D_MODEL;
    const __bf16* wrow = (const __bf16*)wo  + (size_t)(ob + ln) * D_MODEL;

    v8f acc = {};
#pragma unroll 4
    for (int kc = 0; kc < D_MODEL / 32; ++kc) {
        v16bf a = load16(xrow + kc * 32 + half * 16, 8);
        v16bf b = load16(wrow + kc * 32 + half * 16, 8);
        acc = wmma_bf16(a, b, acc);
    }
#pragma unroll
    for (int r = 0; r < 8; ++r) {
        int row = tb + half * 8 + r;
        out[(size_t)row * D_MODEL + ob + ln] = acc[r];
    }
}

// ---------------------------------------------------------------------------
// Host-side launcher
// ---------------------------------------------------------------------------
extern "C" void kernel_launch(void* const* d_in, const int* in_sizes, int n_in,
                              void* d_out, int out_size, void* d_ws,
                              size_t ws_size, hipStream_t stream) {
    (void)in_sizes; (void)n_in; (void)out_size; (void)ws_size;

    const float* hidden = (const float*)d_in[0];
    const float* bias   = (const float*)d_in[1];
    const float* Wq     = (const float*)d_in[2];
    const float* Wk     = (const float*)d_in[3];
    const float* Wv     = (const float*)d_in[4];
    const float* Wo     = (const float*)d_in[5];

    const int NX = B_SZ * T_SEQ * D_MODEL;   // 4,194,304
    const int NW = D_MODEL * D_MODEL;        // 1,048,576

    unsigned short* ws = (unsigned short*)d_ws;
    unsigned short* xb  = ws;                 // bf16 hidden
    unsigned short* wqb = xb + NX;
    unsigned short* wkb = wqb + NW;
    unsigned short* wvb = wkb + NW;
    unsigned short* wob = wvb + NW;
    unsigned short* qo  = wob + NW;           // q [B*T, D]
    unsigned short* ko  = qo + NX;            // k [B*T, D]
    unsigned short* vto = ko + NX;            // v transposed [b,h,d,t]
    unsigned short* ctx = vto + NX;           // attention context [B*T, D]

    cvt_bf16<<<(NX + 255) / 256, 256, 0, stream>>>(hidden, xb, NX);
    cvt_bf16<<<(NW + 255) / 256, 256, 0, stream>>>(Wq, wqb, NW);
    cvt_bf16<<<(NW + 255) / 256, 256, 0, stream>>>(Wk, wkb, NW);
    cvt_bf16<<<(NW + 255) / 256, 256, 0, stream>>>(Wv, wvb, NW);
    cvt_bf16<<<(NW + 255) / 256, 256, 0, stream>>>(Wo, wob, NW);

    qkv_gemm<<<dim3(D_MODEL / 16, (B_SZ * T_SEQ) / 16, 3), 32, 0, stream>>>(
        xb, wqb, wkb, wvb, qo, ko, vto);

    attn_flash<<<dim3(T_SEQ / 16, N_HEADS, B_SZ), 32, 0, stream>>>(
        qo, ko, vto, bias, ctx);

    out_gemm<<<dim3(D_MODEL / 16, (B_SZ * T_SEQ) / 16), 32, 0, stream>>>(
        ctx, wob, (float*)d_out);
}
